// CausalSelfAttention_35227321762258
// MI455X (gfx1250) — matrix operand order repaired and checked by reading, not compiled, not run on hardware
//
#include <hip/hip_runtime.h>
#include <stdint.h>

typedef __bf16 v16bf __attribute__((ext_vector_type(16)));
typedef float  v8f   __attribute__((ext_vector_type(8)));
typedef unsigned int u32x4 __attribute__((ext_vector_type(4)));
typedef unsigned int u32x8 __attribute__((ext_vector_type(8)));

union BFrag { v16bf v; uint4 u[2]; };

__device__ __forceinline__ __bf16 f2bf(float f) {
    unsigned u = __builtin_bit_cast(unsigned, f);
    unsigned r = u + 0x7FFFu + ((u >> 16) & 1u);   // round-to-nearest-even
    unsigned short h = (unsigned short)(r >> 16);
    return __builtin_bit_cast(__bf16, h);
}

// per-lane async global->LDS copy of 16B (tracked by ASYNCcnt)
__device__ __forceinline__ void async_b128(unsigned lds_off, const void* gptr) {
    asm volatile("global_load_async_to_lds_b128 %0, %1, off"
                 :: "v"(lds_off), "v"((unsigned long long)(size_t)gptr)
                 : "memory");
}
__device__ __forceinline__ void wait_async0() {
    asm volatile("s_wait_asynccnt 0x0" ::: "memory");
}

// ---------------- fp32 -> bf16 elementwise ----------------
__global__ void cvt_bf16(const float* __restrict__ in, __bf16* __restrict__ out, int n) {
    int i = blockIdx.x * 256 + threadIdx.x;
    if (i < n) out[i] = f2bf(in[i]);
}

// ---------------- bf16 WMMA GEMM: C[M,N] = A[M,K] * B[K,N] ----------------
// Tile 128x128, K-step 32, 256 threads (8 waves), wave w -> rows [w*16, w*16+16)
// A tile staged by the Tensor Data Mover (wave 0), with LDS row padding 32->40 elems.
__global__ __launch_bounds__(256) void gemm_bf16(const __bf16* __restrict__ A,
                                                 const __bf16* __restrict__ Bm,
                                                 float* __restrict__ C,
                                                 int M, int N, int K) {
    __shared__ alignas(16) __bf16 As[128 * 40];   // [m][k], row padded to 40
    __shared__ alignas(16) __bf16 Bt[128 * 40];   // [n][k] (transposed), row padded to 40
    const int tid = threadIdx.x;
    const int lane = tid & 31, w = tid >> 5;
    const int ln = lane & 15, lh = lane >> 4;
    const int m0 = blockIdx.y * 128, n0 = blockIdx.x * 128;
    v8f acc[8] = {};
    const int bkk = tid >> 3, bnb = (tid & 7) * 16;    // B staging: 16 elems/thread

    // TDM descriptor group 1 (loop-invariant): 2D tile 32(k) x 128(m), 2-byte elems,
    // tensor row stride K, LDS padding: 16 DWORDs data then 4 DWORDs pad (64B+16B = 40 elems)
    u32x8 g1;
    g1[0] = (1u << 16) | (1u << 20) | (3u << 22) | (3u << 25);
    g1[1] = ((unsigned)K & 0xFFFFu) << 16;            // tensor_dim0 lo16
    g1[2] = ((unsigned)K >> 16) | (128u << 16);       // tensor_dim0 hi16 | tensor_dim1 lo16 (=128)
    g1[3] = (32u << 16);                              // tensor_dim1 hi16=0 | tile_dim0=32
    g1[4] = 128u;                                     // tile_dim1=128, tile_dim2=0
    g1[5] = (unsigned)K;                              // tensor_dim0_stride lo32
    g1[6] = 0u;
    g1[7] = 0u;
    const unsigned lds_as = (unsigned)(size_t)&As[0];

    for (int k0 = 0; k0 < K; k0 += 32) {
        // ---- stage A tile via TDM (one wave issues whole 128x32 tile) ----
        if (w == 0) {
            unsigned long long ga = (unsigned long long)(size_t)(A + (size_t)m0 * K + k0);
            u32x4 g0;
            g0[0] = 1u;                               // count=1 (valid descriptor)
            g0[1] = lds_as;                           // LDS byte address
            g0[2] = (unsigned)ga;                     // global addr lo
            g0[3] = (unsigned)(ga >> 32) | 0x80000000u;  // global addr hi | type=2
            asm volatile("tensor_load_to_lds %0, %1" :: "s"(g0), "s"(g1) : "memory");
        }
        // ---- stage B tile transposed: read row-major (coalesced), scatter b16 to [n][k] ----
        {
            union { uint4 u[2]; __bf16 e[16]; } bb;
            const uint4* bs = (const uint4*)(Bm + (size_t)(k0 + bkk) * N + n0 + bnb);
            bb.u[0] = bs[0]; bb.u[1] = bs[1];
            #pragma unroll
            for (int i = 0; i < 16; ++i) Bt[(bnb + i) * 40 + bkk] = bb.e[i];
        }
        // prefetch next K-step B rows into caches while we compute
        if (k0 + 32 < K)
            __builtin_prefetch(Bm + (size_t)(k0 + 32 + bkk) * N + n0 + bnb);
        if (w == 0) __builtin_amdgcn_s_wait_tensorcnt(0);
        __syncthreads();
        // A fragment: lane<16 -> K {0..7,16..23}, lane>=16 -> K {8..15,24..31}
        BFrag af;
        const int mrow = w * 16 + ln, sA = lh * 8;
        af.u[0] = *(const uint4*)(&As[mrow * 40 + sA]);
        af.u[1] = *(const uint4*)(&As[mrow * 40 + sA + 16]);
        #pragma unroll
        for (int j = 0; j < 8; ++j) {
            BFrag bf;   // B fragment: fixed N=j*16+ln, K {lh*8..+7, 16+lh*8..+7}
            const int n = j * 16 + ln, sB = lh * 8;
            bf.u[0] = *(const uint4*)(&Bt[n * 40 + sB]);
            bf.u[1] = *(const uint4*)(&Bt[n * 40 + sB + 16]);
            acc[j] = __builtin_amdgcn_wmma_f32_16x16x32_bf16(false, af.v, false, bf.v,
                                                             (short)0, acc[j], false, false);
        }
        __syncthreads();
    }
    #pragma unroll
    for (int j = 0; j < 8; ++j)
        #pragma unroll
        for (int r = 0; r < 8; ++r)
            C[(size_t)(m0 + w * 16 + r + 8 * lh) * N + (n0 + j * 16 + ln)] = acc[j][r];
}

// ---------------- RoPE + split qkv[B*T,6144] -> Q[b,h,t,d], K/V[b,g,t,d] bf16 ----------------
__global__ void rope_split(const float* __restrict__ qkv,
                           const float* __restrict__ cosb, const float* __restrict__ sinb,
                           __bf16* __restrict__ Q, __bf16* __restrict__ Kc, __bf16* __restrict__ Vc) {
    int i = blockIdx.x * 256 + threadIdx.x;     // over 2*2048*6144
    if (i >= 2 * 2048 * 6144) return;
    int d = i & 127;
    int col = (i >> 7) % 48;                    // g*6 + slot
    int row = i / 6144;                         // b*T + t
    int t = row & 2047, b = row >> 11;
    int g = col / 6, slot = col % 6;
    float v = qkv[i];
    float o;
    if (slot < 5) {                             // q heads + k: apply RoPE
        float rot = (d < 64) ? -qkv[i + 64] : qkv[i - 64];
        o = v * cosb[t * 128 + d] + rot * sinb[t * 128 + d];
    } else o = v;
    __bf16 ob = f2bf(o);
    if (slot < 4)       Q[((size_t)(b * 32 + g * 4 + slot) * 2048 + t) * 128 + d] = ob;
    else if (slot == 4) Kc[((size_t)(b * 8 + g) * 2048 + t) * 128 + d] = ob;
    else                Vc[((size_t)(b * 8 + g) * 2048 + t) * 128 + d] = ob;
}

// ---------------- flash attention: 4 waves/block, 64 queries/block, key step 32 ----------------
__global__ __launch_bounds__(128) void flash_attn(const __bf16* __restrict__ Q,
                                                  const __bf16* __restrict__ Kg,
                                                  const __bf16* __restrict__ Vg,
                                                  __bf16* __restrict__ Y) {
    __shared__ alignas(16) __bf16 Ks[32 * 136];      // [key][d], row padded to 136
    __shared__ alignas(16) __bf16 Vt[128 * 40];      // [d][key] transposed, padded 40
    __shared__ alignas(16) __bf16 Pb[4 * 16 * 40];   // per-wave P staging [16][32] pad 40
    const int tid = threadIdx.x, lane = tid & 31, w = tid >> 5;
    const int ln = lane & 15, lh = lane >> 4;
    const int qb = blockIdx.x * 64;
    const int bh = blockIdx.y, b = bh >> 5, h = bh & 31, g = h >> 2;
    const __bf16* Qp = Q + ((size_t)(b * 32 + h) * 2048 + qb) * 128;
    const __bf16* Kp = Kg + (size_t)(b * 8 + g) * 2048 * 128;
    const __bf16* Vp = Vg + (size_t)(b * 8 + g) * 2048 * 128;
    const int m = w * 16 + ln, sA = lh * 8;
    // Q held in registers as 4 A-fragments (d chunks of 32)
    BFrag qa[4];
    #pragma unroll
    for (int c = 0; c < 4; ++c) {
        qa[c].u[0] = *(const uint4*)(Qp + m * 128 + c * 32 + sA);
        qa[c].u[1] = *(const uint4*)(Qp + m * 128 + c * 32 + sA + 16);
    }
    v8f o[8] = {};
    float mr[8], lr[8];
    #pragma unroll
    for (int r = 0; r < 8; ++r) { mr[r] = -1e30f; lr[r] = 0.0f; }
    const float scale = 0.088388347648318447f;   // 1/sqrt(128)
    const int kend = qb + 64;                    // causal: only key blocks <= diag
    const int krow = tid >> 2, kseg = tid & 3;   // K-tile staging coords (64B per thread)
    const int vkk = tid >> 2, vdb = (tid & 3) * 32;
    for (int kb = 0; kb < kend; kb += 32) {
        {   // stage K tile [32][128] via async LDS loads (4x b128 per lane)
            const __bf16* gk = Kp + (size_t)(kb + krow) * 128 + kseg * 32;
            unsigned lk = (unsigned)(size_t)&Ks[krow * 136 + kseg * 32];
            #pragma unroll
            for (int i = 0; i < 4; ++i)
                async_b128(lk + i * 16, gk + i * 8);
        }
        {   // stage V tile transposed -> Vt[d][key]
            union { uint4 u[4]; __bf16 e[32]; } vb;
            const uint4* s4 = (const uint4*)(Vp + (size_t)(kb + vkk) * 128 + vdb);
            vb.u[0] = s4[0]; vb.u[1] = s4[1]; vb.u[2] = s4[2]; vb.u[3] = s4[3];
            #pragma unroll
            for (int i = 0; i < 32; ++i) Vt[(vdb + i) * 40 + vkk] = vb.e[i];
        }
        if (kb + 32 < kend) {   // prefetch next key block
            __builtin_prefetch(Kp + (size_t)(kb + 32 + krow) * 128 + kseg * 32);
            __builtin_prefetch(Vp + (size_t)(kb + 32 + vkk) * 128 + vdb);
        }
        wait_async0();
        __syncthreads();
        // S = Q * K^T : two 16x16 tiles (key halves), 4 d-chunks each
        v8f s0 = {}, s1 = {};
        #pragma unroll
        for (int c = 0; c < 4; ++c) {
            BFrag b0, b1;
            const int ko = lh * 8 + c * 32;
            b0.u[0] = *(const uint4*)(&Ks[ln * 136 + ko]);
            b0.u[1] = *(const uint4*)(&Ks[ln * 136 + ko + 16]);
            b1.u[0] = *(const uint4*)(&Ks[(16 + ln) * 136 + ko]);
            b1.u[1] = *(const uint4*)(&Ks[(16 + ln) * 136 + ko + 16]);
            s0 = __builtin_amdgcn_wmma_f32_16x16x32_bf16(false, qa[c].v, false, b0.v, (short)0, s0, false, false);
            s1 = __builtin_amdgcn_wmma_f32_16x16x32_bf16(false, qa[c].v, false, b1.v, (short)0, s1, false, false);
        }
        // online softmax; C-fragment row = r + 8*lh, col = lane%16
        __bf16* pw = &Pb[w * 640];
        #pragma unroll
        for (int r = 0; r < 8; ++r) {
            const int qrow = qb + w * 16 + r + 8 * lh;
            float x0 = (kb + ln      <= qrow) ? s0[r] * scale : -1e30f;
            float x1 = (kb + 16 + ln <= qrow) ? s1[r] * scale : -1e30f;
            float mx = fmaxf(x0, x1);
            mx = fmaxf(mx, __shfl_xor(mx, 1));
            mx = fmaxf(mx, __shfl_xor(mx, 2));
            mx = fmaxf(mx, __shfl_xor(mx, 4));
            mx = fmaxf(mx, __shfl_xor(mx, 8));
            const float mn = fmaxf(mr[r], mx);
            const float al = __expf(mr[r] - mn);
            mr[r] = mn;
            const float p0 = __expf(x0 - mn);
            const float p1 = __expf(x1 - mn);
            float rs = p0 + p1;
            rs += __shfl_xor(rs, 1);
            rs += __shfl_xor(rs, 2);
            rs += __shfl_xor(rs, 4);
            rs += __shfl_xor(rs, 8);
            lr[r] = lr[r] * al + rs;
            #pragma unroll
            for (int j = 0; j < 8; ++j) o[j][r] *= al;
            pw[(r + 8 * lh) * 40 + ln] = f2bf(p0);
            pw[(r + 8 * lh) * 40 + 16 + ln] = f2bf(p1);
        }
        // P as one 16x32 A-fragment; O += P * V (8 d-tiles)
        BFrag pa;
        pa.u[0] = *(const uint4*)(&Pb[w * 640 + ln * 40 + sA]);
        pa.u[1] = *(const uint4*)(&Pb[w * 640 + ln * 40 + sA + 16]);
        #pragma unroll
        for (int j = 0; j < 8; ++j) {
            BFrag vf;
            vf.u[0] = *(const uint4*)(&Vt[(j * 16 + ln) * 40 + lh * 8]);
            vf.u[1] = *(const uint4*)(&Vt[(j * 16 + ln) * 40 + lh * 8 + 16]);
            o[j] = __builtin_amdgcn_wmma_f32_16x16x32_bf16(false, pa.v, false, vf.v, (short)0, o[j], false, false);
        }
        __syncthreads();
    }
    // finalize: O /= l, write y[b*T + t, h*128 + d] as bf16 for the proj GEMM
    #pragma unroll
    for (int r = 0; r < 8; ++r) {
        const float inv = 1.0f / lr[r];
        const size_t yrow = (size_t)(b * 2048 + qb + w * 16 + r + 8 * lh) * 4096 + h * 128;
        #pragma unroll
        for (int j = 0; j < 8; ++j)
            Y[yrow + j * 16 + ln] = f2bf(o[j][r] * inv);
    }
}

extern "C" void kernel_launch(void* const* d_in, const int* in_sizes, int n_in,
                              void* d_out, int out_size, void* d_ws, size_t ws_size,
                              hipStream_t stream) {
    (void)in_sizes; (void)n_in; (void)out_size; (void)ws_size;
    const float* x     = (const float*)d_in[0];
    const float* cosb  = (const float*)d_in[1];
    const float* sinb  = (const float*)d_in[2];
    const float* Wqkv  = (const float*)d_in[3];
    const float* Wproj = (const float*)d_in[4];
    float* out = (float*)d_out;

    char* ws = (char*)d_ws;
    const size_t NX  = 2ull * 2048 * 4096;   // 16,777,216
    const size_t NWQ = 4096ull * 6144;       // 25,165,824
    const size_t NWP = 4096ull * 4096;       // 16,777,216
    // layout (with aliasing; total ~208 MB):
    __bf16* xbf  = (__bf16*)(ws);                        // [0, 32MB)   ; reused as ybf
    __bf16* wqbf = (__bf16*)(ws + (33554432ull));        // [32, 80MB)  ; reused as Q/K/V
    __bf16* wpbf = (__bf16*)(ws + (83886080ull));        // [80, 112MB)
    float*  qkvf = (float*) (ws + (117440512ull));       // [112, 208MB)
    __bf16* Qbf  = wqbf;                                 // 32MB
    __bf16* Kbf  = (__bf16*)(ws + (67108864ull));        // 8MB
    __bf16* Vbf  = (__bf16*)(ws + (75497472ull));        // 8MB
    __bf16* ybf  = xbf;

    cvt_bf16<<<(int)(NX  / 256), 256, 0, stream>>>(x,     xbf,  (int)NX);
    cvt_bf16<<<(int)(NWQ / 256), 256, 0, stream>>>(Wqkv,  wqbf, (int)NWQ);
    cvt_bf16<<<(int)(NWP / 256), 256, 0, stream>>>(Wproj, wpbf, (int)NWP);

    // qkv = x @ Wqkv  (M=4096, N=6144, K=4096)
    gemm_bf16<<<dim3(6144 / 128, 4096 / 128), 256, 0, stream>>>(xbf, wqbf, qkvf, 4096, 6144, 4096);

    // RoPE + split into head-major bf16 Q/K/V (overwrites Wqkv_bf region; safe)
    rope_split<<<(int)(NWQ / 256), 256, 0, stream>>>(qkvf, cosb, sinb, Qbf, Kbf, Vbf);

    // flash attention: grid (query blocks of 64, B*H heads)
    flash_attn<<<dim3(2048 / 64, 64), 128, 0, stream>>>(Qbf, Kbf, Vbf, ybf);

    // out = y @ Wproj  (M=4096, N=4096, K=4096)
    gemm_bf16<<<dim3(4096 / 128, 4096 / 128), 256, 0, stream>>>(ybf, wpbf, out, 4096, 4096, 4096);
}
